// GraphAttention_41575283425673
// MI455X (gfx1250) — matrix operand order repaired and checked by reading, not compiled, not run on hardware
//
#include <hip/hip_runtime.h>
#include <hip/hip_fp16.h>

typedef __attribute__((ext_vector_type(16))) _Float16 v16h;
typedef __attribute__((ext_vector_type(8)))  float    v8f;
typedef __attribute__((ext_vector_type(2)))  float    v2f;

#define N_NODES 8192
#define F_IN    128
#define F_OUT   64
#define LEAKY   0.2f

__device__ __forceinline__ float leaky_relu(float s) { return s > 0.f ? s : LEAKY * s; }

// ---------------------------------------------------------------------------
// Kernel 1: seq_fts = x @ W (WMMA f16->f32), emit sf16 (f16 copy) + f1/f2 rows
// One wave computes a 16x64 tile; 8 waves/block -> 128 rows/block, 64 blocks.
// ---------------------------------------------------------------------------
__global__ __launch_bounds__(256) void gat_gemm1(
    const float* __restrict__ x, const float* __restrict__ W,
    const float* __restrict__ a1, const float* __restrict__ a2,
    _Float16* __restrict__ sf16, float* __restrict__ f1, float* __restrict__ f2)
{
    __shared__ __align__(64) _Float16 Wl[F_IN * F_OUT];   // 16 KB
    const int tid = threadIdx.x;
    for (int i = tid; i < F_IN * F_OUT; i += 256) Wl[i] = (_Float16)W[i];
    __syncthreads();

    const int wave = tid >> 5;
    const int lane = tid & 31;
    const int nn   = lane & 15;
    const int hi   = lane >> 4;
    const int m0   = blockIdx.x * 128 + wave * 16;
    const int mA   = m0 + nn;                 // A-matrix row owned by this lane

    v8f zero = {};
    v8f acc[4] = { zero, zero, zero, zero };

    #pragma unroll
    for (int kt = 0; kt < 4; ++kt) {          // K = 128 in chunks of 32
        v16h a;
        #pragma unroll
        for (int p = 0; p < 8; ++p) {
            const int kk = ((p & 3) * 2) + ((p >> 2) * 16) + hi * 8;
            v2f xv = *(const v2f*)(x + (size_t)mA * F_IN + kt * 32 + kk);
            a[2 * p]     = (_Float16)xv.x;
            a[2 * p + 1] = (_Float16)xv.y;
        }
        #pragma unroll
        for (int nt = 0; nt < 4; ++nt) {      // N = 64 in tiles of 16
            v16h b;
            #pragma unroll
            for (int s = 0; s < 16; ++s) b[s] = Wl[(kt * 32 + (s < 8 ? s : s + 8) + hi * 8) * F_OUT + nt * 16 + nn];
            acc[nt] = __builtin_amdgcn_wmma_f32_16x16x32_f16(
                false, a, false, b, (short)0, acc[nt], false, false);
        }
    }

    float a1v[4], a2v[4];
    #pragma unroll
    for (int nt = 0; nt < 4; ++nt) { a1v[nt] = a1[nt * 16 + nn]; a2v[nt] = a2[nt * 16 + nn]; }

    #pragma unroll
    for (int r = 0; r < 8; ++r) {
        const int m = m0 + r + hi * 8;
        float p1 = 0.f, p2 = 0.f;
        #pragma unroll
        for (int nt = 0; nt < 4; ++nt) {
            float v = acc[nt][r];
            p1 += v * a1v[nt];
            p2 += v * a2v[nt];
            sf16[(size_t)m * F_OUT + nt * 16 + nn] = (_Float16)v;
        }
        #pragma unroll
        for (int mask = 1; mask < 16; mask <<= 1) {   // reduce within 16-lane half
            p1 += __shfl_xor(p1, mask, 32);
            p2 += __shfl_xor(p2, mask, 32);
        }
        if (nn == 0) { f1[m] = p1; f2[m] = p2; }
    }
}

// ---------------------------------------------------------------------------
// Kernel 2: maxf2 = max_j f2[j]  (softmax row-max is g(f1[i]+maxf2), g monotone)
// ---------------------------------------------------------------------------
__global__ __launch_bounds__(256) void gat_maxf2(const float* __restrict__ f2,
                                                 float* __restrict__ mx)
{
    __shared__ float red[256];
    const int tid = threadIdx.x;
    float m = -3.0e38f;
    for (int i = tid; i < N_NODES; i += 256) m = fmaxf(m, f2[i]);
    red[tid] = m;
    __syncthreads();
    for (int s = 128; s > 0; s >>= 1) {
        if (tid < s) red[tid] = fmaxf(red[tid], red[tid + s]);
        __syncthreads();
    }
    if (tid == 0) *mx = red[0];
}

// ---------------------------------------------------------------------------
// Kernel 3: streaming softmax-weighted GEMM (flash style).
// Each wave: 16 output rows. Block streams seq_fts in 32-row chunks through a
// double-buffered LDS tile filled with global_load_async_to_lds_b128.
// Row sums via an extra WMMA against an all-ones B.
// ---------------------------------------------------------------------------
__global__ __launch_bounds__(256) void gat_attn(
    const _Float16* __restrict__ sf16, const float* __restrict__ f1,
    const float* __restrict__ f2, const float* __restrict__ maxf2p,
    const float* __restrict__ bvec, float* __restrict__ out)
{
    __shared__ __align__(64) _Float16 Bbuf[2 * 32 * F_OUT];   // 8 KB double buffer
    const int tid  = threadIdx.x;
    const int wave = tid >> 5;
    const int lane = tid & 31;
    const int nn   = lane & 15;
    const int hi   = lane >> 4;
    const int m0   = blockIdx.x * 128 + wave * 16;
    const int mA   = m0 + nn;

    const float f1m = f1[mA];
    const float mF2 = *maxf2p;
    const float mi  = leaky_relu(f1m + mF2);   // exact row max of scores

    v16h ones;
    #pragma unroll
    for (int i = 0; i < 16; ++i) ones[i] = (_Float16)1.0f;

    v8f zero = {};
    v8f acc[4] = { zero, zero, zero, zero };
    v8f accS = zero;

    const unsigned ldsBase = (unsigned)(unsigned long long)(const void*)&Bbuf[0];
    const int NCHUNK = N_NODES / 32;           // 256

    // prefetch chunk 0 into buffer 0 (wave 0 only; 8 x b128 x 32 lanes = 4 KB)
    if (tid < 32) {
        const char* g = (const char*)sf16 + lane * 16;
        const unsigned l = ldsBase + (unsigned)(lane * 16);
        #pragma unroll
        for (int i = 0; i < 8; ++i) {
            unsigned loff = l + (unsigned)(i * 512);
            unsigned long long ga = (unsigned long long)(g + i * 512);
            asm volatile("global_load_async_to_lds_b128 %0, %1, off"
                         :: "v"(loff), "v"(ga) : "memory");
        }
    }

    for (int c = 0; c < NCHUNK; ++c) {
        const int bi = c & 1;
        if (tid < 32) {
            if (c + 1 < NCHUNK) {
                const char* g = (const char*)sf16 + (size_t)(c + 1) * 4096 + lane * 16;
                const unsigned l = ldsBase + (unsigned)(((bi ^ 1) * 4096) + lane * 16);
                #pragma unroll
                for (int i = 0; i < 8; ++i) {
                    unsigned loff = l + (unsigned)(i * 512);
                    unsigned long long ga = (unsigned long long)(g + i * 512);
                    asm volatile("global_load_async_to_lds_b128 %0, %1, off"
                                 :: "v"(loff), "v"(ga) : "memory");
                }
                asm volatile("s_wait_asynccnt 8" ::: "memory"); // chunk c landed
            } else {
                asm volatile("s_wait_asynccnt 0" ::: "memory");
            }
        }
        __syncthreads();   // buffer bi visible to all waves

        // A fragment: w = exp(leaky(f1[i]+f2[j]) - m_i), 16 rows x 32 cols, f16
        v16h a;
        #pragma unroll
        for (int p = 0; p < 8; ++p) {
            const int kk = ((p & 3) * 2) + ((p >> 2) * 16) + hi * 8;
            v2f fp = *(const v2f*)(f2 + c * 32 + kk);
            float e0 = __expf(leaky_relu(f1m + fp.x) - mi);
            float e1 = __expf(leaky_relu(f1m + fp.y) - mi);
            a[2 * p]     = (_Float16)e0;
            a[2 * p + 1] = (_Float16)e1;
        }

        const _Float16* bb = &Bbuf[bi * 32 * F_OUT];
        #pragma unroll
        for (int nt = 0; nt < 4; ++nt) {
            v16h b;
            #pragma unroll
            for (int s = 0; s < 16; ++s) b[s] = bb[((s < 8 ? s : s + 8) + hi * 8) * F_OUT + nt * 16 + nn];
            acc[nt] = __builtin_amdgcn_wmma_f32_16x16x32_f16(
                false, a, false, b, (short)0, acc[nt], false, false);
        }
        accS = __builtin_amdgcn_wmma_f32_16x16x32_f16(
            false, a, false, ones, (short)0, accS, false, false);

        __syncthreads();   // done with buffer bi before it is refilled
    }

    float bv[4];
    #pragma unroll
    for (int nt = 0; nt < 4; ++nt) bv[nt] = bvec[nt * 16 + nn];

    #pragma unroll
    for (int r = 0; r < 8; ++r) {
        const int m = m0 + r + hi * 8;
        const float inv = 1.0f / accS[r];
        #pragma unroll
        for (int nt = 0; nt < 4; ++nt) {
            float v = acc[nt][r] * inv + bv[nt];
            out[(size_t)m * F_OUT + nt * 16 + nn] = fmaxf(v, 0.f);
        }
    }
}

// ---------------------------------------------------------------------------
extern "C" void kernel_launch(void* const* d_in, const int* in_sizes, int n_in,
                              void* d_out, int out_size, void* d_ws, size_t ws_size,
                              hipStream_t stream)
{
    (void)in_sizes; (void)n_in; (void)out_size; (void)ws_size;

    const float* x  = (const float*)d_in[0];
    const float* W  = (const float*)d_in[1];
    const float* a1 = (const float*)d_in[2];
    const float* a2 = (const float*)d_in[3];
    const float* b  = (const float*)d_in[4];
    float* out = (float*)d_out;

    _Float16* sf16 = (_Float16*)d_ws;                                  // 1 MB
    float* f1 = (float*)((char*)d_ws + (size_t)N_NODES * F_OUT * sizeof(_Float16));
    float* f2 = f1 + N_NODES;
    float* mx = f2 + N_NODES;

    gat_gemm1<<<N_NODES / 128, 256, 0, stream>>>(x, W, a1, a2, sf16, f1, f2);
    gat_maxf2<<<1, 256, 0, stream>>>(f2, mx);
    gat_attn<<<N_NODES / 128, 256, 0, stream>>>(sf16, f1, f2, mx, b, out);
}
